// GAT_41558103556513
// MI455X (gfx1250) — compile-verified, matrix operand order checked by reading, not run on hardware
//
#include <hip/hip_runtime.h>
#include <cstdint>

// ---------------------------------------------------------------------------
// Types for CDNA5 WMMA (wave32): 16x16x32 bf16 -> f32
// ---------------------------------------------------------------------------
typedef unsigned short bf16_t;
typedef __attribute__((ext_vector_type(16))) __bf16 v16bf;
typedef __attribute__((ext_vector_type(8)))  float  v8f;
typedef __attribute__((ext_vector_type(4)))  unsigned int u32x4;
typedef __attribute__((ext_vector_type(8)))  int i32x8;
typedef __attribute__((ext_vector_type(4)))  int i32x4;

union FragU { v16bf v; uint4 u[2]; };

__device__ __forceinline__ bf16_t f2bf(float f) {
  unsigned u = __float_as_uint(f);
  u += 0x7FFFu + ((u >> 16) & 1u);   // round-to-nearest-even
  return (bf16_t)(u >> 16);
}

// order-preserving float<->int mapping for atomic segment-max
__device__ __forceinline__ int f2ord(float f) {
  int i = __float_as_int(f);
  return (i >= 0) ? i : (int)(i ^ 0x7FFFFFFF);
}
__device__ __forceinline__ float ord2f(int i) {
  return __int_as_float((i >= 0) ? i : (int)(i ^ 0x7FFFFFFF));
}

__device__ __forceinline__ void atomicAddF(float* p, float v) {
  __hip_atomic_fetch_add(p, v, __ATOMIC_RELAXED, __HIP_MEMORY_SCOPE_AGENT);
}

#define HEADS 4
#define DHEAD 64
#define HD    256   // HEADS*DHEAD
#define KMAX  512
#define LDSPAD 8    // +8 bf16 elements per column: breaks 16-way LDS bank conflict

// ---------------------------------------------------------------------------
// f32 -> bf16 elementwise convert
// ---------------------------------------------------------------------------
__global__ void k_f32_to_bf16(const float* __restrict__ in, bf16_t* __restrict__ out,
                              long long n) {
  long long i = (long long)blockIdx.x * blockDim.x + threadIdx.x;
  long long stride = (long long)gridDim.x * blockDim.x;
  for (; i < n; i += stride) out[i] = f2bf(in[i]);
}

// W [K, Nout] f32 -> Wt [Nout, K] bf16  (so B-fragments load contiguously)
__global__ void k_transpose_w(const float* __restrict__ W, bf16_t* __restrict__ Wt,
                              int K, int Nout) {
  long long n = (long long)K * Nout;
  long long i = (long long)blockIdx.x * blockDim.x + threadIdx.x;
  long long stride = (long long)gridDim.x * blockDim.x;
  for (; i < n; i += stride) {
    int k = (int)(i / Nout), c = (int)(i % Nout);
    Wt[(long long)c * K + k] = f2bf(W[i]);
  }
}

__global__ void k_fill_i32(int* __restrict__ p, int v, long long n) {
  long long i = (long long)blockIdx.x * blockDim.x + threadIdx.x;
  long long stride = (long long)gridDim.x * blockDim.x;
  for (; i < n; i += stride) p[i] = v;
}

// ---------------------------------------------------------------------------
// WMMA GEMM:  C[M,Nout] f32 = A[M,K] bf16  x  Bt[Nout,K]^T bf16
//
// Block = 256 threads (8 waves) computes a 128(M) x 64(N) tile.
// The shared 64-col x K strip of B is staged once into LDS by the Tensor
// Data Mover (TDM descriptor with pad_enable: +4 DWORDs every K elements,
// giving a K+8 element padded column stride -> conflict-free ds_load_b128).
// K loop per wave: 2 global b128 (A) + 8 ds_load_b128 (B) + 4 WMMA,
// A fragment reused across the 4 N-tiles.
// ---------------------------------------------------------------------------
__global__ void k_gemm_bf16_wmma(const bf16_t* __restrict__ A,
                                 const bf16_t* __restrict__ Bt,
                                 float* __restrict__ C,
                                 int M, int K, int Nout) {
  __shared__ bf16_t ldsB[64 * (KMAX + LDSPAD)];
  const int nTiles = Nout >> 6;                 // 64-wide N strips
  const int bm = blockIdx.x / nTiles;
  const int bn = blockIdx.x - bm * nTiles;
  const int ldb = K + LDSPAD;                   // padded LDS column stride (elems)

  // ---- stage B strip (64 cols x K) into LDS ----
#if __has_builtin(__builtin_amdgcn_tensor_load_to_lds) && __has_builtin(__builtin_amdgcn_s_wait_tensorcnt)
  if (threadIdx.x < 32) {                       // one wave issues the DMA
    unsigned lds_base = (unsigned)(unsigned long long)(const void*)ldsB;
    unsigned long long gaddr =
        (unsigned long long)(const void*)(Bt + (size_t)bn * 64 * K);
    // D# group 0: count=1 | lds_addr | global_addr[56:0] | type=2
    u32x4 g0;
    g0[0] = 1u;
    g0[1] = lds_base;
    g0[2] = (unsigned)(gaddr & 0xFFFFFFFFu);
    g0[3] = (unsigned)((gaddr >> 32) & 0x01FFFFFFu) | (2u << 30);
    // D# group 1: data_size=2B, pad_enable, pad every K elems (K/2 dwords)
    // by 4 dwords (=LDSPAD bf16); dims: dim0=K, dim1=Nout; tile 64 x K.
    const int pi = (K == 512) ? 7 : 6;          // 2^(pi+1) dwords between pads
    const int pa = 3;                           // pa+1 = 4 dwords of padding
    i32x8 g1;
    g1[0] = (1 << 16) | (1 << 20) | (pi << 22) | (pa << 25);
    g1[1] = (K & 0xFFFF) << 16;                 // tensor_dim0[15:0]
    g1[2] = ((unsigned)K >> 16) | ((Nout & 0xFFFF) << 16);   // dim0 hi | dim1 lo
    g1[3] = ((unsigned)Nout >> 16) | ((K & 0xFFFF) << 16);   // dim1 hi | tile_dim0
    g1[4] = 64;                                 // tile_dim1=64, tile_dim2=0
    g1[5] = K;                                  // tensor_dim0_stride[31:0]
    g1[6] = 0;
    g1[7] = 0;
    i32x4 z4 = (i32x4){0, 0, 0, 0};             // groups 2/3 unused (2D tensor)
    i32x8 z8 = (i32x8){0, 0, 0, 0, 0, 0, 0, 0};
    __builtin_amdgcn_tensor_load_to_lds(g0, g1, z4, z4, z8, 0);
    __builtin_amdgcn_s_wait_tensorcnt(0);
  }
#else
  {
    // cooperative fallback: global b128 -> ds_store_b128 with padded stride
    const int chunksPerCol = K / 8;             // 8 bf16 per 16B chunk
    const int total = 64 * chunksPerCol;
    for (int c = threadIdx.x; c < total; c += blockDim.x) {
      int col = c / chunksPerCol, kc = c - col * chunksPerCol;
      uint4 v = *(const uint4*)(Bt + ((size_t)(bn * 64 + col)) * K + kc * 8);
      *(uint4*)(ldsB + (size_t)col * ldb + kc * 8) = v;
    }
  }
#endif
  __syncthreads();

  const int wave = threadIdx.x >> 5;
  const int lane = threadIdx.x & 31;
  const int mTile = bm * 8 + wave;              // wave-uniform
  if (mTile * 16 >= M) return;                  // no barriers past this point

  const int half = lane >> 4;
  const int lidx = lane & 15;

  int arIdx = mTile * 16 + lidx; if (arIdx >= M) arIdx = M - 1;
  const bf16_t* arow = A + (size_t)arIdx * K;
  const bf16_t* lb0 = ldsB + (size_t)(0 * 16 + lidx) * ldb;
  const bf16_t* lb1 = ldsB + (size_t)(1 * 16 + lidx) * ldb;
  const bf16_t* lb2 = ldsB + (size_t)(2 * 16 + lidx) * ldb;
  const bf16_t* lb3 = ldsB + (size_t)(3 * 16 + lidx) * ldb;

  v8f acc[4];
#pragma unroll
  for (int j = 0; j < 4; ++j) acc[j] = (v8f){0.f,0.f,0.f,0.f,0.f,0.f,0.f,0.f};

  for (int k0 = 0; k0 < K; k0 += 32) {
    FragU a;
    a.u[0] = *(const uint4*)(arow + k0 + half * 8);       // K = k0 + half*8 ..+8
    a.u[1] = *(const uint4*)(arow + k0 + 16 + half * 8);  // K = k0+16+half*8 ..+8
    FragU b0, b1, b2, b3;
    b0.u[0] = *(const uint4*)(lb0 + k0 + half * 8);
    b0.u[1] = *(const uint4*)(lb0 + k0 + 16 + half * 8);
    b1.u[0] = *(const uint4*)(lb1 + k0 + half * 8);
    b1.u[1] = *(const uint4*)(lb1 + k0 + 16 + half * 8);
    b2.u[0] = *(const uint4*)(lb2 + k0 + half * 8);
    b2.u[1] = *(const uint4*)(lb2 + k0 + 16 + half * 8);
    b3.u[0] = *(const uint4*)(lb3 + k0 + half * 8);
    b3.u[1] = *(const uint4*)(lb3 + k0 + 16 + half * 8);
    acc[0] = __builtin_amdgcn_wmma_f32_16x16x32_bf16(
        false, a.v, false, b0.v, (short)0, acc[0], false, false);
    acc[1] = __builtin_amdgcn_wmma_f32_16x16x32_bf16(
        false, a.v, false, b1.v, (short)0, acc[1], false, false);
    acc[2] = __builtin_amdgcn_wmma_f32_16x16x32_bf16(
        false, a.v, false, b2.v, (short)0, acc[2], false, false);
    acc[3] = __builtin_amdgcn_wmma_f32_16x16x32_bf16(
        false, a.v, false, b3.v, (short)0, acc[3], false, false);
  }

  // C/D layout: VGPR r -> row = r + half*8, col = lane&15 (per 16x16 tile)
#pragma unroll
  for (int j = 0; j < 4; ++j) {
    int col = bn * 64 + j * 16 + lidx;
#pragma unroll
    for (int r = 0; r < 8; ++r) {
      int row = mTile * 16 + half * 8 + r;
      if (row < M) C[(size_t)row * Nout + col] = acc[j][r];
    }
  }
}

// ---------------------------------------------------------------------------
// Attention logits: el[n,h] = <feat[n,h,:], al[h,:]>, er likewise
// ---------------------------------------------------------------------------
__global__ void k_attn_logits(const float* __restrict__ feat,
                              const float* __restrict__ al,
                              const float* __restrict__ ar,
                              float* __restrict__ el, float* __restrict__ er,
                              int N) {
  int i = blockIdx.x * blockDim.x + threadIdx.x;   // i = n*H + h
  if (i >= N * HEADS) return;
  int n = i >> 2, h = i & 3;
  const float* f = feat + (size_t)n * HD + h * DHEAD;
  const float* a = al + h * DHEAD;
  const float* b = ar + h * DHEAD;
  float sl = 0.f, sr = 0.f;
#pragma unroll 8
  for (int d = 0; d < DHEAD; ++d) { float v = f[d]; sl += v * a[d]; sr += v * b[d]; }
  el[i] = sl; er[i] = sr;
}

__device__ __forceinline__ float lrelu(float v) { return v > 0.f ? v : 0.2f * v; }

// segment max (ordered-int atomicMax)
__global__ void k_edge_max(const float* __restrict__ el, const float* __restrict__ er,
                           const int* __restrict__ src, const int* __restrict__ dst,
                           int* __restrict__ mi, int E) {
  int i = blockIdx.x * blockDim.x + threadIdx.x;   // i = e*H + h
  if (i >= E * HEADS) return;
  int e = i >> 2, h = i & 3;
  float v = lrelu(el[src[e] * HEADS + h] + er[dst[e] * HEADS + h]);
  atomicMax(mi + dst[e] * HEADS + h, f2ord(v));
}

// a = exp(e - m[dst]); s[dst] += a
__global__ void k_edge_exp(const float* __restrict__ el, const float* __restrict__ er,
                           const int* __restrict__ src, const int* __restrict__ dst,
                           const int* __restrict__ mi, float* __restrict__ s,
                           float* __restrict__ ae, int E) {
  int i = blockIdx.x * blockDim.x + threadIdx.x;
  if (i >= E * HEADS) return;
  int e = i >> 2, h = i & 3;
  int d = dst[e];
  float v = lrelu(el[src[e] * HEADS + h] + er[d * HEADS + h]);
  float m = ord2f(mi[d * HEADS + h]);
  float a = __expf(v - m);
  ae[i] = a;
  atomicAddF(s + d * HEADS + h, a);
}

__global__ void k_edge_norm(float* __restrict__ ae, const float* __restrict__ s,
                            const int* __restrict__ dst, int E) {
  int i = blockIdx.x * blockDim.x + threadIdx.x;
  if (i >= E * HEADS) return;
  int e = i >> 2, h = i & 3;
  float sv = s[dst[e] * HEADS + h];
  ae[i] = ae[i] / (sv > 0.f ? sv : 1.f);
}

// agg[dst, h, d] += a[e,h] * feat[src, h, d]  (feat fits in 192MB L2)
__global__ void k_edge_agg(const float* __restrict__ ae, const float* __restrict__ feat,
                           const int* __restrict__ src, const int* __restrict__ dst,
                           float* __restrict__ agg, long long E) {
  long long total = E * HD;
  long long i = (long long)blockIdx.x * blockDim.x + threadIdx.x;
  long long stride = (long long)gridDim.x * blockDim.x;
  for (; i < total; i += stride) {
    long long e = i >> 8;          // HD == 256
    int t = (int)(i & 255);
    int sn = src[e], dn = dst[e];
    if (t == 0) __builtin_prefetch(feat + (size_t)sn * HD, 0, 0); // global_prefetch_b8
    float w = ae[e * HEADS + (t >> 6)];
    atomicAddF(agg + (size_t)dn * HD + t, w * feat[(size_t)sn * HD + t]);
  }
}

// ELU then bf16 (input to next layer's WMMA GEMM)
__global__ void k_elu_bf16(const float* __restrict__ in, bf16_t* __restrict__ out,
                           long long n) {
  long long i = (long long)blockIdx.x * blockDim.x + threadIdx.x;
  long long stride = (long long)gridDim.x * blockDim.x;
  for (; i < n; i += stride) {
    float v = in[i];
    out[i] = f2bf(v > 0.f ? v : (__expf(v) - 1.f));
  }
}

// mean over heads: out[n,d] = 0.25 * sum_h agg[n,h,d]
__global__ void k_head_mean(const float* __restrict__ agg, float* __restrict__ out, int N) {
  int i = blockIdx.x * blockDim.x + threadIdx.x;   // i = n*64 + d
  if (i >= N * DHEAD) return;
  int n = i >> 6, d = i & 63;
  const float* p = agg + (size_t)n * HD + d;
  out[i] = 0.25f * (p[0] + p[DHEAD] + p[2 * DHEAD] + p[3 * DHEAD]);
}

// ---------------------------------------------------------------------------
// Host orchestration
// ---------------------------------------------------------------------------
static void run_attention(const float* feat, const float* al, const float* ar,
                          const int* src, const int* dst,
                          float* el, float* er, int* mi, float* s, float* ae, float* agg,
                          int N, int E, hipStream_t stream) {
  const int T = 256;
  (void)hipMemsetAsync(s, 0, (size_t)N * HEADS * sizeof(float), stream);
  (void)hipMemsetAsync(agg, 0, (size_t)N * HD * sizeof(float), stream);
  k_fill_i32<<<1024, T, 0, stream>>>(mi, (int)0x80000000, (long long)N * HEADS);
  k_attn_logits<<<(N * HEADS + T - 1) / T, T, 0, stream>>>(feat, al, ar, el, er, N);
  k_edge_max<<<(E * HEADS + T - 1) / T, T, 0, stream>>>(el, er, src, dst, mi, E);
  k_edge_exp<<<(E * HEADS + T - 1) / T, T, 0, stream>>>(el, er, src, dst, mi, s, ae, E);
  k_edge_norm<<<(E * HEADS + T - 1) / T, T, 0, stream>>>(ae, s, dst, E);
  k_edge_agg<<<8192, T, 0, stream>>>(ae, feat, src, dst, agg, (long long)E);
}

extern "C" void kernel_launch(void* const* d_in, const int* in_sizes, int n_in,
                              void* d_out, int out_size, void* d_ws, size_t ws_size,
                              hipStream_t stream) {
  const float* x   = (const float*)d_in[0];
  const int*   src = (const int*)d_in[1];
  const int*   dst = (const int*)d_in[2];
  const float* W1  = (const float*)d_in[3];
  const float* al1 = (const float*)d_in[4];
  const float* ar1 = (const float*)d_in[5];
  const float* W2  = (const float*)d_in[6];
  const float* al2 = (const float*)d_in[7];
  const float* ar2 = (const float*)d_in[8];
  float* out = (float*)d_out;

  const int F_IN = 512;
  const int N = in_sizes[0] / F_IN;
  const int E = in_sizes[1];

  // ---- workspace carve-up (256B aligned) ----
  char* ws = (char*)d_ws;
  size_t off = 0;
  auto take = [&](size_t bytes) -> void* {
    void* p = ws + off;
    off += (bytes + 255) & ~(size_t)255;
    return p;
  };
  bf16_t* Ab   = (bf16_t*)take((size_t)N * F_IN * sizeof(bf16_t)); // x_bf16, later h_bf16
  bf16_t* Wt   = (bf16_t*)take((size_t)F_IN * HD * sizeof(bf16_t)); // W1t, later W2t
  float*  feat = (float*) take((size_t)N * HD * sizeof(float));
  float*  el   = (float*) take((size_t)N * HEADS * sizeof(float));
  float*  er   = (float*) take((size_t)N * HEADS * sizeof(float));
  int*    mi   = (int*)   take((size_t)N * HEADS * sizeof(int));
  float*  s    = (float*) take((size_t)N * HEADS * sizeof(float));
  float*  ae   = (float*) take((size_t)E * HEADS * sizeof(float));
  float*  agg  = (float*) take((size_t)N * HD * sizeof(float));
  (void)ws_size; (void)n_in; (void)out_size;

  const int T = 256;

  // ================= Layer 1 =================
  k_f32_to_bf16<<<4096, T, 0, stream>>>(x, Ab, (long long)N * F_IN);
  k_transpose_w<<<1024, T, 0, stream>>>(W1, Wt, F_IN, HD);
  {
    int mTiles = (N + 15) / 16, nTiles = HD / 64;
    int blocks = ((mTiles + 7) / 8) * nTiles;
    k_gemm_bf16_wmma<<<blocks, 256, 0, stream>>>(Ab, Wt, feat, N, F_IN, HD);
  }
  run_attention(feat, al1, ar1, src, dst, el, er, mi, s, ae, agg, N, E, stream);

  // ELU + convert -> next layer input (bf16), overwrites Ab
  k_elu_bf16<<<4096, T, 0, stream>>>(agg, Ab, (long long)N * HD);

  // ================= Layer 2 =================
  k_transpose_w<<<1024, T, 0, stream>>>(W2, Wt, HD, HD);
  {
    int mTiles = (N + 15) / 16, nTiles = HD / 64;
    int blocks = ((mTiles + 7) / 8) * nTiles;
    k_gemm_bf16_wmma<<<blocks, 256, 0, stream>>>(Ab, Wt, feat, N, HD, HD);
  }
  run_attention(feat, al2, ar2, src, dst, el, er, mi, s, ae, agg, N, E, stream);

  // mean over heads -> [N, 64] f32
  k_head_mean<<<(N * DHEAD + T - 1) / T, T, 0, stream>>>(agg, out, N);
}
